// MultiHeadAttention_85504208929166
// MI455X (gfx1250) — matrix operand order repaired and checked by reading, not compiled, not run on hardware
//
#include <hip/hip_runtime.h>
#include <hip/hip_bf16.h>
#include <stdint.h>

// Problem constants (match reference)
#define EMB   1024
#define HEADS 16
#define HD    64
#define SEQ   2048
#define BATCH 2
#define ROWS  (BATCH * SEQ)   // 4096

typedef __attribute__((ext_vector_type(16))) __bf16   v16bf;
typedef __attribute__((ext_vector_type(8)))  float    v8f;
typedef __attribute__((ext_vector_type(4)))  uint32_t u32x4;
typedef __attribute__((ext_vector_type(4)))  float    f32x4;

union Frag {
  v16bf          v;
  u32x4          q[2];
  uint32_t       w[8];
  unsigned short h[16];
};

union Pack16 {
  u32x4          q[2];
  unsigned short h[16];
};

__device__ __forceinline__ unsigned short f32_bf16(float f) {
  uint32_t u = __builtin_bit_cast(uint32_t, f);
  u += 0x7FFFu + ((u >> 16) & 1u);   // round-to-nearest-even
  return (unsigned short)(u >> 16);
}

// ---------------------------------------------------------------------------
// Tiled GEMM: C[rowbase..+64, colbase..+64] = A[4096 x 1024] @ W[1024 x 1024]
// A is fp32 (converted to bf16 on LDS stage) or bf16 (already converted).
// OUTMODE 0: fp32 C + bias -> out[r*EMB + c]        (final output projection)
// OUTMODE 1: bf16 C + bias -> out[((b*H+h)*S+s)*64+d] (QKV head-major layout)
// Block: 128 threads = 4 waves; wave w computes rows [16w,16w+16) x 64 cols.
// ---------------------------------------------------------------------------
template <bool A_IS_BF16, int OUTMODE>
__global__ __launch_bounds__(128)
void mha_gemm(const void* __restrict__ Ap, const float* __restrict__ W,
              const float* __restrict__ bias, void* __restrict__ outp)
{
  __shared__ __align__(16) unsigned short lA[64 * 32];  // [m][k] bf16
  __shared__ __align__(16) unsigned short lB[32 * 64];  // [k][n] bf16

  const int tid  = threadIdx.x;
  const int wave = tid >> 5;
  const int lane = tid & 31;
  const int half = lane >> 4;   // 0: lanes 0-15, 1: lanes 16-31
  const int l16  = lane & 15;
  const int rowbase = blockIdx.x * 64;
  const int colbase = blockIdx.y * 64;

  v8f acc[4] = {};

  const int arow = tid >> 1;            // 0..63
  const int acol = (tid & 1) * 16;      // 0,16
  const int brow = tid >> 2;            // 0..31
  const int bcol = (tid & 3) * 16;      // 0,16,32,48

  for (int k0 = 0; k0 < EMB; k0 += 32) {
    // ---- stage A tile (64x32), converting fp32 -> bf16 if needed ----
    Pack16 pa;
    if (A_IS_BF16) {
      const u32x4* g = (const u32x4*)((const unsigned short*)Ap +
                        (size_t)(rowbase + arow) * EMB + k0 + acol);
      pa.q[0] = g[0];
      pa.q[1] = g[1];
    } else {
      const f32x4* g = (const f32x4*)((const float*)Ap +
                        (size_t)(rowbase + arow) * EMB + k0 + acol);
#pragma unroll
      for (int j = 0; j < 4; ++j) {
        f32x4 f = g[j];
#pragma unroll
        for (int e = 0; e < 4; ++e) pa.h[4 * j + e] = f32_bf16(f[e]);
      }
    }
    *(u32x4*)&lA[arow * 32 + acol]     = pa.q[0];
    *(u32x4*)&lA[arow * 32 + acol + 8] = pa.q[1];

    // ---- stage B tile (32x64), fp32 -> bf16 ----
    Pack16 pb;
    {
      const f32x4* g = (const f32x4*)(W + (size_t)(k0 + brow) * EMB +
                                      colbase + bcol);
#pragma unroll
      for (int j = 0; j < 4; ++j) {
        f32x4 f = g[j];
#pragma unroll
        for (int e = 0; e < 4; ++e) pb.h[4 * j + e] = f32_bf16(f[e]);
      }
    }
    *(u32x4*)&lB[brow * 64 + bcol]     = pb.q[0];
    *(u32x4*)&lB[brow * 64 + bcol + 8] = pb.q[1];

    __syncthreads();

    // ---- A fragment: 16x32, lane holds row M=l16; K split by half ----
    Frag a;
    {
      const unsigned short* ap = &lA[(16 * wave + l16) * 32 + half * 16];
      a.q[0] = *(const u32x4*)ap;          // K = half*16 .. +7
      a.q[1] = *(const u32x4*)(ap + 8);    // K = half*16 + 8 .. +15
    }

#pragma unroll
    for (int nt = 0; nt < 4; ++nt) {
      // B fragment: lane holds col n = nt*16+l16; VGPR i holds K=half*16+2i,+1
      Frag b;
#pragma unroll
      for (int i = 0; i < 16; ++i)
        b.h[i] = lB[(half * 16 + i) * 64 + nt * 16 + l16];
      acc[nt] = __builtin_amdgcn_wmma_f32_16x16x32_bf16(
          false, a.v, false, b.v, (short)0, acc[nt], false, false);
    }
    __syncthreads();
  }

  // ---- epilogue ----
  float bv[4];
#pragma unroll
  for (int nt = 0; nt < 4; ++nt) bv[nt] = bias[colbase + nt * 16 + l16];

#pragma unroll
  for (int nt = 0; nt < 4; ++nt) {
#pragma unroll
    for (int g = 0; g < 8; ++g) {
      float v = acc[nt][g] + bv[nt];
      int r = rowbase + 16 * wave + g + half * 8;  // C layout: VGPR g = row g / g+8
      int c = colbase + nt * 16 + l16;
      if (OUTMODE == 0) {
        ((float*)outp)[(size_t)r * EMB + c] = v;
      } else {
        int bb = r / SEQ, s = r % SEQ;
        int hh = c / HD, d = c % HD;
        ((unsigned short*)outp)[(((size_t)(bb * HEADS + hh)) * SEQ + s) * HD + d] =
            f32_bf16(v);
      }
    }
  }
}

// ---------------------------------------------------------------------------
// Flash attention: grid (S/64, B*HEADS), 128 threads = 4 waves.
// Wave w owns q-rows [s0+16w, s0+16w+16). Streams K/V in 64-key tiles through
// LDS, keeps scores in registers, online softmax, accumulates O in fp32.
// Writes Z (pre-output-projection) as bf16 in [b, s, h*64+d] layout.
// ---------------------------------------------------------------------------
__global__ __launch_bounds__(128)
void mha_flash(const unsigned short* __restrict__ Qh,
               const unsigned short* __restrict__ Kh,
               const unsigned short* __restrict__ Vh,
               unsigned short* __restrict__ Z)
{
  __shared__ __align__(16) unsigned short lK[64 * 64];      // [key][d]
  __shared__ __align__(16) unsigned short lV[64 * 64];      // [key][d]
  __shared__ __align__(16) unsigned short lP[4][16 * 64];   // per-wave P

  const int tid  = threadIdx.x;
  const int wave = tid >> 5;
  const int lane = tid & 31;
  const int half = lane >> 4;
  const int l16  = lane & 15;
  const int bh   = blockIdx.y;
  const int s0   = blockIdx.x * 64;

  const unsigned short* Qb = Qh + (size_t)bh * SEQ * HD;
  const unsigned short* Kb = Kh + (size_t)bh * SEQ * HD;
  const unsigned short* Vb = Vh + (size_t)bh * SEQ * HD;

  // Q fragments for this wave's 16 rows, two 16x32 slices over d
  Frag qf[2];
  {
    const unsigned short* qp = Qb + (size_t)(s0 + 16 * wave + l16) * HD;
    qf[0].q[0] = *(const u32x4*)(qp + half * 8);
    qf[0].q[1] = *(const u32x4*)(qp + 16 + half * 8);
    qf[1].q[0] = *(const u32x4*)(qp + 32 + half * 8);
    qf[1].q[1] = *(const u32x4*)(qp + 48 + half * 8);
  }

  float m_run[8], l_run[8];
#pragma unroll
  for (int g = 0; g < 8; ++g) { m_run[g] = -1e30f; l_run[g] = 0.0f; }
  v8f acc[4] = {};

  const int cr = tid >> 1;          // 0..63 (key row for staging)
  const int cc = (tid & 1) * 32;    // 0 or 32 (d chunk, 32 bf16 = 64B)

  for (int kt = 0; kt < SEQ / 64; ++kt) {
    // ---- stage K and V 64x64 tiles into LDS (vectorized 128b copies) ----
    {
      const u32x4* gk = (const u32x4*)(Kb + (size_t)(kt * 64 + cr) * HD + cc);
      const u32x4* gv = (const u32x4*)(Vb + (size_t)(kt * 64 + cr) * HD + cc);
      u32x4* dk = (u32x4*)&lK[cr * 64 + cc];
      u32x4* dv = (u32x4*)&lV[cr * 64 + cc];
#pragma unroll
      for (int j = 0; j < 4; ++j) { dk[j] = gk[j]; dv[j] = gv[j]; }
    }
    if (kt + 1 < SEQ / 64) {
      __builtin_prefetch(Kb + (size_t)((kt + 1) * 64 + cr) * HD + cc, 0, 1);
      __builtin_prefetch(Vb + (size_t)((kt + 1) * 64 + cr) * HD + cc, 0, 1);
    }
    __syncthreads();

    // ---- scores S = Q @ K^T  (16q x 64keys per wave) ----
    v8f sc[4] = {};
#pragma unroll
    for (int nt = 0; nt < 4; ++nt) {
#pragma unroll
      for (int ds = 0; ds < 2; ++ds) {
        // B frag: lane col n = key nt*16+l16; K(=d) = ds*32 + half*8 + 0..7 and
        // ds*32 + 16 + half*8 + 0..7 -> two 128b LDS reads.
        Frag kb;
        const unsigned short* kp = &lK[(nt * 16 + l16) * 64 + ds * 32 + half * 8];
        kb.q[0] = *(const u32x4*)kp;
        kb.q[1] = *(const u32x4*)(kp + 16);
        sc[nt] = __builtin_amdgcn_wmma_f32_16x16x32_bf16(
            false, qf[ds].v, false, kb.v, (short)0, sc[nt], false, false);
      }
      sc[nt] *= 0.125f;   // 1/sqrt(d), d = 64
    }

    // ---- online softmax (row = VGPR index g, within 16-lane halves) ----
    float mnew[8], alpha[8];
#pragma unroll
    for (int g = 0; g < 8; ++g) {
      float v = fmaxf(fmaxf(sc[0][g], sc[1][g]), fmaxf(sc[2][g], sc[3][g]));
#pragma unroll
      for (int msk = 1; msk <= 8; msk <<= 1)
        v = fmaxf(v, __shfl_xor(v, msk, 32));
      mnew[g]  = fmaxf(m_run[g], v);
      alpha[g] = __expf(m_run[g] - mnew[g]);
      m_run[g] = mnew[g];
    }
#pragma unroll
    for (int nt = 0; nt < 4; ++nt)
#pragma unroll
      for (int g = 0; g < 8; ++g)
        sc[nt][g] = __expf(sc[nt][g] - mnew[g]);
#pragma unroll
    for (int g = 0; g < 8; ++g) {
      float rs = sc[0][g] + sc[1][g] + sc[2][g] + sc[3][g];
#pragma unroll
      for (int msk = 1; msk <= 8; msk <<= 1)
        rs += __shfl_xor(rs, msk, 32);
      l_run[g] = l_run[g] * alpha[g] + rs;
    }
#pragma unroll
    for (int nt = 0; nt < 4; ++nt)
#pragma unroll
      for (int g = 0; g < 8; ++g)
        acc[nt][g] *= alpha[g];

    // ---- P: C-layout -> A-layout via per-wave LDS round trip ----
    unsigned short* P = lP[wave];
#pragma unroll
    for (int nt = 0; nt < 4; ++nt)
#pragma unroll
      for (int g = 0; g < 8; ++g)
        P[(g + half * 8) * 64 + nt * 16 + l16] = f32_bf16(sc[nt][g]);
    // Same-wave DS ops are in-order; barrier only the compiler.
    asm volatile("" ::: "memory");

    Frag pf[2];
#pragma unroll
    for (int ks = 0; ks < 2; ++ks) {
      const unsigned short* pp = &P[l16 * 64 + ks * 32 + half * 16];
      pf[ks].q[0] = *(const u32x4*)pp;
      pf[ks].q[1] = *(const u32x4*)(pp + 8);
    }

    // ---- O += P @ V ----
#pragma unroll
    for (int dt = 0; dt < 4; ++dt) {
#pragma unroll
      for (int ks = 0; ks < 2; ++ks) {
        // B frag: lane col n = d dt*16+l16; K(=key) = ks*32 + half*16 + i
        Frag vb;
#pragma unroll
        for (int i = 0; i < 16; ++i)
          vb.h[i] = lV[(ks * 32 + half * 16 + i) * 64 + dt * 16 + l16];
        acc[dt] = __builtin_amdgcn_wmma_f32_16x16x32_bf16(
            false, pf[ks].v, false, vb.v, (short)0, acc[dt], false, false);
      }
    }
    __syncthreads();
  }

  // ---- finalize: O /= l, write Z bf16 in [b, s, h*64+d] ----
  const int b = bh / HEADS, h = bh % HEADS;
  float invl[8];
#pragma unroll
  for (int g = 0; g < 8; ++g) invl[g] = 1.0f / l_run[g];
#pragma unroll
  for (int dt = 0; dt < 4; ++dt) {
#pragma unroll
    for (int g = 0; g < 8; ++g) {
      int s = s0 + 16 * wave + g + half * 8;
      int c = h * HD + dt * 16 + l16;
      Z[((size_t)(b * SEQ + s)) * EMB + c] = f32_bf16(acc[dt][g] * invl[g]);
    }
  }
}

// ---------------------------------------------------------------------------
// Launch: 3 projection GEMMs -> flash attention -> output GEMM.
// Workspace: Qh/Kh/Vh/Z bf16, 8 MiB each (32 MiB total).
// ---------------------------------------------------------------------------
extern "C" void kernel_launch(void* const* d_in, const int* in_sizes, int n_in,
                              void* d_out, int out_size, void* d_ws, size_t ws_size,
                              hipStream_t stream) {
  (void)in_sizes; (void)n_in; (void)out_size; (void)ws_size;

  const float* query = (const float*)d_in[0];
  const float* key   = (const float*)d_in[1];
  const float* value = (const float*)d_in[2];
  const float* Wq    = (const float*)d_in[3];
  const float* bq    = (const float*)d_in[4];
  const float* Wk    = (const float*)d_in[5];
  const float* bk    = (const float*)d_in[6];
  const float* Wv    = (const float*)d_in[7];
  const float* bv    = (const float*)d_in[8];
  const float* Wo    = (const float*)d_in[9];
  const float* bo    = (const float*)d_in[10];
  float* out = (float*)d_out;

  char* ws = (char*)d_ws;
  unsigned short* Qh = (unsigned short*)(ws + (size_t)0);
  unsigned short* Kh = (unsigned short*)(ws + ((size_t)8 << 20));
  unsigned short* Vh = (unsigned short*)(ws + ((size_t)16 << 20));
  unsigned short* Zb = (unsigned short*)(ws + ((size_t)24 << 20));

  dim3 ggrid(ROWS / 64, EMB / 64);  // (64, 16)
  mha_gemm<false, 1><<<ggrid, 128, 0, stream>>>(query, Wq, bq, Qh);
  mha_gemm<false, 1><<<ggrid, 128, 0, stream>>>(key,   Wk, bk, Kh);
  mha_gemm<false, 1><<<ggrid, 128, 0, stream>>>(value, Wv, bv, Vh);

  mha_flash<<<dim3(SEQ / 64, BATCH * HEADS), 128, 0, stream>>>(Qh, Kh, Vh, Zb);

  mha_gemm<true, 0><<<ggrid, 128, 0, stream>>>(Zb, Wo, bo, out);
}
